// MoE_29781303230537
// MI455X (gfx1250) — matrix operand order repaired and checked by reading, not compiled, not run on hardware
//
#include <hip/hip_runtime.h>
#include <math.h>

// ---------------- problem constants ----------------
#define D_MODEL 1024
#define D_FF    4096
#define NUM_E   8
#define BATCH   4
#define SEQ     2048
#define TOKENS  (BATCH * SEQ)      // 8192
#define TCHUNK  2048               // token chunk for FFN intermediates

typedef unsigned short u16;
typedef __attribute__((ext_vector_type(16))) __bf16 v16bf;
typedef __attribute__((ext_vector_type(8)))  float  v8f;
typedef __attribute__((ext_vector_type(4)))  unsigned int v4u;
typedef __attribute__((ext_vector_type(8)))  int v8i;
typedef __attribute__((ext_vector_type(4)))  int v4i;

// ---- TDM availability / arity hedge ----
#if defined(__has_builtin)
#if __has_builtin(__builtin_amdgcn_tensor_load_to_lds)
#define MOE_HAS_TDM 1
#endif
#endif
#ifndef MOE_HAS_TDM
#define MOE_HAS_TDM 0
#endif

#if MOE_HAS_TDM
#if __has_include(<hip/amd_detail/amd_gfx1250_TDM.h>)
// therock-10.0 headers -> clang-23 6-arg builtin
#define MOE_TDM_LOAD(g0, g1) \
    __builtin_amdgcn_tensor_load_to_lds((g0), (g1), (v4i){0,0,0,0}, (v4i){0,0,0,0}, \
                                        (v8i){0,0,0,0,0,0,0,0}, 0)
#else
// ROCm 7.2 clang-22 5-arg builtin (probe-verified signature)
#define MOE_TDM_LOAD(g0, g1) \
    __builtin_amdgcn_tensor_load_to_lds((g0), (g1), (v4i){0,0,0,0}, (v4i){0,0,0,0}, 0)
#endif
#endif

__device__ __forceinline__ u16 f32_to_bf16(float f) {
    unsigned int u = __float_as_uint(f);
    u += 0x7FFFu + ((u >> 16) & 1u);   // round-to-nearest-even
    return (u16)(u >> 16);
}

// ---------------- misc elementwise kernels ----------------
__global__ void moe_zero_f32(float* __restrict__ p, long n) {
    long i = (long)blockIdx.x * blockDim.x + threadIdx.x;
    long stride = (long)gridDim.x * blockDim.x;
    for (; i < n; i += stride) p[i] = 0.0f;
}

__global__ void moe_cvt_bf16(const float* __restrict__ s, u16* __restrict__ d, long n) {
    long i = (long)blockIdx.x * blockDim.x + threadIdx.x;
    long stride = (long)gridDim.x * blockDim.x;
    for (; i < n; i += stride) d[i] = f32_to_bf16(s[i]);
}

// ---------------- gating: logits ----------------
// one block (8 waves) per token; wave w computes expert w's logit
__global__ __launch_bounds__(256) void moe_gating(
    const float* __restrict__ x, const float* __restrict__ gw,
    const float* __restrict__ gb, float* __restrict__ logits) {
    int t = blockIdx.x;
    int wave = threadIdx.x >> 5;
    int lane = threadIdx.x & 31;
    const float* xr = x + (size_t)t * D_MODEL;
    float acc = 0.0f;
    for (int i = lane; i < D_MODEL; i += 32)
        acc += xr[i] * gw[i * NUM_E + wave];
    #pragma unroll
    for (int off = 16; off > 0; off >>= 1)
        acc += __shfl_xor(acc, off, 32);
    if (lane == 0) logits[t * NUM_E + wave] = acc + gb[wave];
}

// ---------------- gating: softmax over SEQUENCE axis ----------------
// one block per (b,e) column of length SEQ (=2048 = 8*256); in-place
__global__ __launch_bounds__(256) void moe_seq_softmax(float* __restrict__ logits) {
    int b = blockIdx.x / NUM_E;
    int e = blockIdx.x % NUM_E;
    int tid = threadIdx.x;
    __shared__ float red[256];
    float v[8];
    float mx = -3.4e38f;
    #pragma unroll
    for (int p = 0; p < 8; ++p) {
        int s = tid + p * 256;
        v[p] = logits[((size_t)b * SEQ + s) * NUM_E + e];
        mx = fmaxf(mx, v[p]);
    }
    red[tid] = mx; __syncthreads();
    for (int off = 128; off > 0; off >>= 1) {
        if (tid < off) red[tid] = fmaxf(red[tid], red[tid + off]);
        __syncthreads();
    }
    mx = red[0]; __syncthreads();
    float sum = 0.0f;
    #pragma unroll
    for (int p = 0; p < 8; ++p) { v[p] = expf(v[p] - mx); sum += v[p]; }
    red[tid] = sum; __syncthreads();
    for (int off = 128; off > 0; off >>= 1) {
        if (tid < off) red[tid] += red[tid + off];
        __syncthreads();
    }
    float inv = 1.0f / red[0];
    #pragma unroll
    for (int p = 0; p < 8; ++p)
        logits[((size_t)b * SEQ + tid + p * 256) * NUM_E + e] = v[p] * inv;
}

// ---------------- top-2 -> coefficients for experts 0 and 1 ----------------
__global__ __launch_bounds__(256) void moe_topk_coeff(
    const float* __restrict__ w, float* __restrict__ coef) {
    int t = blockIdx.x * 256 + threadIdx.x;
    if (t >= TOKENS) return;
    float vv[NUM_E];
    #pragma unroll
    for (int e = 0; e < NUM_E; ++e) vv[e] = w[t * NUM_E + e];
    int i1 = 0; float v1 = vv[0];
    #pragma unroll
    for (int e = 1; e < NUM_E; ++e) if (vv[e] > v1) { v1 = vv[e]; i1 = e; }
    int i2 = -1; float v2 = -3.4e38f;
    #pragma unroll
    for (int e = 0; e < NUM_E; ++e) if (e != i1 && vv[e] > v2) { v2 = vv[e]; i2 = e; }
    float c0 = (i1 == 0 ? v1 : 0.0f) + (i2 == 0 ? v2 : 0.0f);
    float c1 = (i1 == 1 ? v1 : 0.0f) + (i2 == 1 ? v2 : 0.0f);
    coef[t * 2 + 0] = c0;
    coef[t * 2 + 1] = c1;
}

// ---------------- transpose + convert fp32 [R,C] -> bf16 [C,R] ----------------
__global__ __launch_bounds__(256) void moe_transpose_cvt(
    const float* __restrict__ src, u16* __restrict__ dst, int R, int C) {
    __shared__ float tile[32][33];
    int c0 = blockIdx.x * 32, r0 = blockIdx.y * 32;
    int tx = threadIdx.x, ty = threadIdx.y;     // block (32,8)
    #pragma unroll
    for (int i = 0; i < 32; i += 8)
        tile[ty + i][tx] = src[(size_t)(r0 + ty + i) * C + (c0 + tx)];
    __syncthreads();
    #pragma unroll
    for (int i = 0; i < 32; i += 8)
        dst[(size_t)(c0 + ty + i) * R + (r0 + tx)] = f32_to_bf16(tile[tx][ty + i]);
}

// ---------------- bf16 WMMA GEMM: C[M,N] = A[M,K] * Bt[N,K]^T + bias ----------------
// macro tile 128x128, 8 waves of 64x32, K-step 64.
// Tiles staged into double-buffered LDS by the Tensor Data Mover (per-wave
// tensor_load_to_lds of a 16x64 slice, TDM pad -> 144B row stride), so the
// fetch of tile k+1 overlaps WMMA on tile k; sync = s_wait_tensorcnt + barrier.
// MODE 0: out = bf16(gelu_exact(c + bias))  -> outH
// MODE 1: out = c + bias (fp32)             -> outF
#define BM 128
#define BN 128
#define BK 64
#define LDT 72   // 64 + 8 ushort pad -> 144B row stride, conflict-free b128 reads

#if MOE_HAS_TDM
// Issue one TDM load of a 16-row x 64-col bf16 slice into padded LDS rows.
__device__ __forceinline__ void moe_tdm_tile16x64(
    const u16* gsrc, unsigned ldsAddr, int K, int rowsTot) {
    unsigned long long ga = (unsigned long long)(uintptr_t)gsrc;
    v4u g0;
    g0[0] = 1u;                                        // count=1, user mode
    g0[1] = ldsAddr;                                   // LDS byte address
    g0[2] = (unsigned)ga;                              // global addr [31:0]
    g0[3] = ((unsigned)(ga >> 32) & 0x01FFFFFFu) | 0x80000000u; // [56:32] | type=2
    v8i g1;
    // workgroup_mask=0 | data_size=1 (2B) | pad_enable | pad_interval=4 (32 dw)
    // | pad_amount=3 (4 dwords = 16B)
    g1[0] = (int)((1u << 16) | (1u << 20) | (4u << 22) | (3u << 25));
    g1[1] = (int)(((unsigned)K & 0xFFFFu) << 16);                       // tensor_dim0 lo
    g1[2] = (int)(((unsigned)K >> 16) | (((unsigned)rowsTot & 0xFFFFu) << 16));
    g1[3] = (int)((((unsigned)rowsTot >> 16) & 0xFFFFu) | (64u << 16)); // tile_dim0=64
    g1[4] = 16;                                        // tile_dim1=16 rows, tile_dim2=0
    g1[5] = K;                                         // tensor_dim0_stride lo
    g1[6] = 0;
    g1[7] = 0;
    MOE_TDM_LOAD(g0, g1);
}
#endif

template <int MODE>
__global__ __launch_bounds__(256) void moe_gemm_bf16(
    const u16* __restrict__ A, const u16* __restrict__ Bt,
    const float* __restrict__ bias, u16* __restrict__ outH,
    float* __restrict__ outF, int M, int N, int K) {
    __shared__ __align__(16) u16 Asl[2][BM * LDT];
    __shared__ __align__(16) u16 Bsl[2][BN * LDT];

    const int tid  = threadIdx.x;
    const int wave = tid >> 5;
    const int lane = tid & 31;
    const int half = lane >> 4;    // 0 or 1
    const int r    = lane & 15;
    const int wm   = (wave >> 2) * 64;   // wave M offset inside tile (0/64)
    const int wn   = (wave & 3) * 32;    // wave N offset inside tile (0..96)
    const int mBase = blockIdx.y * BM;
    const int nBase = blockIdx.x * BN;

    v8f acc[4][2] = {};

    const int tcol = (tid & 7) * 8;      // K offset within tile (elements)
    const int trow = tid >> 3;           // 0..31
    const u16* Ag = A  + (size_t)(mBase + trow) * K + tcol;
    const u16* Bg = Bt + (size_t)(nBase + trow) * K + tcol;

#if MOE_HAS_TDM
    // per-wave global slice bases (uniform within the wave)
    const u16* AgW = A  + (size_t)(mBase + wave * 16) * K;
    const u16* BgW = Bt + (size_t)(nBase + wave * 16) * K;

    // prologue: stage tile 0 into buffer 0
    moe_tdm_tile16x64(AgW, (unsigned)(uintptr_t)&Asl[0][wave * 16 * LDT], K, M);
    moe_tdm_tile16x64(BgW, (unsigned)(uintptr_t)&Bsl[0][wave * 16 * LDT], K, N);
    __builtin_amdgcn_s_wait_tensorcnt(0);
    __syncthreads();
#endif

    int buf = 0;
    for (int kt = 0; kt < K; kt += BK) {
#if MOE_HAS_TDM
        // overlap: DMA next tile into the other buffer while we compute
        if (kt + BK < K) {
            moe_tdm_tile16x64(AgW + (kt + BK),
                              (unsigned)(uintptr_t)&Asl[buf ^ 1][wave * 16 * LDT], K, M);
            moe_tdm_tile16x64(BgW + (kt + BK),
                              (unsigned)(uintptr_t)&Bsl[buf ^ 1][wave * 16 * LDT], K, N);
        }
        if (kt + 2 * BK < K) {   // pre-warm GL2 for the tile after next
            __builtin_prefetch(Ag + kt + 2 * BK, 0, 1);
            __builtin_prefetch(Bg + kt + 2 * BK, 0, 1);
        }
#else
        // fallback: cooperative VGPR staging (b128 global -> b128 ds)
        #pragma unroll
        for (int p = 0; p < 4; ++p) {
            *(uint4*)&Asl[buf][(trow + 32 * p) * LDT + tcol] =
                *(const uint4*)(Ag + (size_t)(32 * p) * K + kt);
            *(uint4*)&Bsl[buf][(trow + 32 * p) * LDT + tcol] =
                *(const uint4*)(Bg + (size_t)(32 * p) * K + kt);
        }
        if (kt + BK < K) {
            __builtin_prefetch(Ag + kt + BK, 0, 1);
            __builtin_prefetch(Bg + kt + BK, 0, 1);
        }
        __syncthreads();
#endif

        #pragma unroll
        for (int ks = 0; ks < BK; ks += 32) {
            v16bf a[4], b[2];
            // A fragment (ISA 16-bit A 16x32): half0 -> K[0..7],[16..23]; half1 -> +8
            #pragma unroll
            for (int i = 0; i < 4; ++i) {
                uint4* ap = (uint4*)&a[i];
                const u16* base = &Asl[buf][(wm + i * 16 + r) * LDT + ks + half * 8];
                ap[0] = *(const uint4*)(base);
                ap[1] = *(const uint4*)(base + 16);
            }
            // B fragment: same K order as A: half0 -> K[0..7],[16..23]; half1 -> +8
            #pragma unroll
            for (int j = 0; j < 2; ++j) {
                uint4* bp = (uint4*)&b[j];
                const u16* base = &Bsl[buf][(wn + j * 16 + r) * LDT + ks + half * 8];
                bp[0] = *(const uint4*)(base);
                bp[1] = *(const uint4*)(base + 16);
            }
            #pragma unroll
            for (int i = 0; i < 4; ++i)
                #pragma unroll
                for (int j = 0; j < 2; ++j)
                    acc[i][j] = __builtin_amdgcn_wmma_f32_16x16x32_bf16(
                        false, a[i], false, b[j], (short)0, acc[i][j], false, false);
        }

#if MOE_HAS_TDM
        __builtin_amdgcn_s_wait_tensorcnt(0);  // this wave's DMAs into buf^1 done
        __syncthreads();                       // everyone done reading buf + all DMAs done
        buf ^= 1;
#else
        __syncthreads();
#endif
    }

    // epilogue; C layout: element = (M = v + half*8, N = r) within each 16x16
    #pragma unroll
    for (int i = 0; i < 4; ++i) {
        int row0 = mBase + wm + i * 16 + half * 8;
        #pragma unroll
        for (int j = 0; j < 2; ++j) {
            int col = nBase + wn + j * 16 + r;
            float bv = bias[col];
            #pragma unroll
            for (int v = 0; v < 8; ++v) {
                float cv = acc[i][j][v] + bv;
                int row = row0 + v;
                if (MODE == 0) {
                    float g = 0.5f * cv * (1.0f + erff(cv * 0.70710678118654752f));
                    outH[(size_t)row * N + col] = f32_to_bf16(g);
                } else {
                    outF[(size_t)row * N + col] = cv;
                }
            }
        }
    }
}

// ---------------- row softmax over D + weighted accumulate ----------------
__global__ __launch_bounds__(256) void moe_softmax_accum(
    const float* __restrict__ Y, const float* __restrict__ coef,
    float* __restrict__ out, int tokenBase, int expert) {
    int t = blockIdx.x;
    int g = tokenBase + t;
    int tid = threadIdx.x;
    float c = coef[g * 2 + expert];
    const float* y = Y + (size_t)t * D_MODEL;
    float* o = out + (size_t)g * D_MODEL;
    __shared__ float red[256];
    float v[4];
    float mx = -3.4e38f;
    #pragma unroll
    for (int p = 0; p < 4; ++p) {
        v[p] = y[tid + p * 256];
        mx = fmaxf(mx, v[p]);
    }
    red[tid] = mx; __syncthreads();
    for (int off = 128; off > 0; off >>= 1) {
        if (tid < off) red[tid] = fmaxf(red[tid], red[tid + off]);
        __syncthreads();
    }
    mx = red[0]; __syncthreads();
    float sum = 0.0f;
    #pragma unroll
    for (int p = 0; p < 4; ++p) { v[p] = expf(v[p] - mx); sum += v[p]; }
    red[tid] = sum; __syncthreads();
    for (int off = 128; off > 0; off >>= 1) {
        if (tid < off) red[tid] += red[tid + off];
        __syncthreads();
    }
    float scale = c / red[0];
    #pragma unroll
    for (int p = 0; p < 4; ++p)
        o[tid + p * 256] += scale * v[p];
}

// ---------------- launcher ----------------
extern "C" void kernel_launch(void* const* d_in, const int* in_sizes, int n_in,
                              void* d_out, int out_size, void* d_ws, size_t ws_size,
                              hipStream_t stream) {
    const float* x      = (const float*)d_in[0];
    const float* gate_w = (const float*)d_in[1];
    const float* gate_b = (const float*)d_in[2];
    const float* w1     = (const float*)d_in[3];
    const float* b1     = (const float*)d_in[4];
    const float* w2     = (const float*)d_in[5];
    const float* b2     = (const float*)d_in[6];
    float* out = (float*)d_out;
    char*  ws  = (char*)d_ws;

    // workspace layout (~57 MB)
    float* logits = (float*)(ws + 0x0000000);          // T*E f32   (256KB)
    float* coef   = (float*)(ws + 0x0040000);          // T*2 f32   (64KB)
    u16*   Xbf    = (u16*)  (ws + 0x0100000);          // T*D bf16  (16MB)
    u16*   W1T    = (u16*)  (ws + 0x1100000);          // F*D bf16  (8MB)  [N=F, K=D]
    u16*   W2T    = (u16*)  (ws + 0x1900000);          // D*F bf16  (8MB)  [N=D, K=F]
    u16*   Hbf    = (u16*)  (ws + 0x2100000);          // TC*F bf16 (16MB)
    float* Yf     = (float*)(ws + 0x3100000);          // TC*D f32  (8MB)

    // 1) zero output (accumulated into later)
    moe_zero_f32<<<4096, 256, 0, stream>>>(out, (long)TOKENS * D_MODEL);

    // 2) gating -> seq-axis softmax -> top-2 coefficients
    moe_gating<<<TOKENS, 256, 0, stream>>>(x, gate_w, gate_b, logits);
    moe_seq_softmax<<<BATCH * NUM_E, 256, 0, stream>>>(logits);
    moe_topk_coeff<<<TOKENS / 256, 256, 0, stream>>>(logits, coef);

    // 3) x -> bf16
    moe_cvt_bf16<<<4096, 256, 0, stream>>>(x, Xbf, (long)TOKENS * D_MODEL);

    // 4) experts 0 and 1 (reference quirk: only these are ever computed)
    for (int e = 0; e < 2; ++e) {
        // w1[e]: [D,F] -> W1T [F,D] bf16 ; w2[e]: [F,D] -> W2T [D,F] bf16
        moe_transpose_cvt<<<dim3(D_FF / 32, D_MODEL / 32), dim3(32, 8), 0, stream>>>(
            w1 + (size_t)e * D_MODEL * D_FF, W1T, D_MODEL, D_FF);
        moe_transpose_cvt<<<dim3(D_MODEL / 32, D_FF / 32), dim3(32, 8), 0, stream>>>(
            w2 + (size_t)e * D_FF * D_MODEL, W2T, D_FF, D_MODEL);

        for (int ch = 0; ch < TOKENS / TCHUNK; ++ch) {
            const u16* Ach = Xbf + (size_t)ch * TCHUNK * D_MODEL;
            // H = gelu(X @ W1 + b1) -> bf16
            moe_gemm_bf16<0><<<dim3(D_FF / BN, TCHUNK / BM), 256, 0, stream>>>(
                Ach, W1T, b1 + (size_t)e * D_FF, Hbf, nullptr,
                TCHUNK, D_FF, D_MODEL);
            // Y = H @ W2 + b2 -> f32
            moe_gemm_bf16<1><<<dim3(D_MODEL / BN, TCHUNK / BM), 256, 0, stream>>>(
                Hbf, W2T, b2 + (size_t)e * D_MODEL, nullptr, Yf,
                TCHUNK, D_MODEL, D_FF);
            // out += c_e * softmax_row(Y)
            moe_softmax_accum<<<TCHUNK, 256, 0, stream>>>(Yf, coef, out, ch * TCHUNK, e);
        }
    }
}
